// MultiHeadDiffAttention_59442347376853
// MI455X (gfx1250) — matrix operand order repaired and checked by reading, not compiled, not run on hardware
//
#include <hip/hip_runtime.h>
#include <math.h>

// Problem dims (fixed by the reference)
#define BB  2
#define SSQ 2048
#define EEE 1024
#define HHH 8
#define DHD 128
#define FFF 64

typedef __attribute__((ext_vector_type(16))) _Float16 v16h;
typedef __attribute__((ext_vector_type(8)))  _Float16 v8h;
typedef __attribute__((ext_vector_type(8)))  float    v8f;

union frag_u { v16h v; v8h h[2]; };

static __device__ inline v16h frag_from2(const _Float16* p0, const _Float16* p1) {
  frag_u u;
  u.h[0] = *(const v8h*)p0;
  u.h[1] = *(const v8h*)p1;
  return u.v;
}

static __device__ inline v8f wmma_f16(v16h a, v16h b, v8f c) {
  // D = A*B + C, fp32 accum.  (neg_a, A, neg_b, B, c_mod, C, reuse_a, reuse_b)
  return __builtin_amdgcn_wmma_f32_16x16x32_f16(false, a, false, b, (short)0, c,
                                                false, false);
}

// ---------------------------------------------------------------------------
// f32 -> f16 cast, 8 elems/thread (n8 = n/8)
// ---------------------------------------------------------------------------
__global__ void cast_f32_to_f16_kernel(const float* __restrict__ src,
                                       _Float16* __restrict__ dst, int n8) {
  int i = blockIdx.x * blockDim.x + threadIdx.x;
  if (i >= n8) return;
  const float4* s4 = (const float4*)src;
  float4 a = s4[2 * i], b = s4[2 * i + 1];
  v8h o;
  o[0] = (_Float16)a.x; o[1] = (_Float16)a.y;
  o[2] = (_Float16)a.z; o[3] = (_Float16)a.w;
  o[4] = (_Float16)b.x; o[5] = (_Float16)b.y;
  o[6] = (_Float16)b.z; o[7] = (_Float16)b.w;
  *(v8h*)(dst + 8 * i) = o;
}

// ---------------------------------------------------------------------------
// C[m,n] = sum_k A[m,k] * W[n,k]   (X @ W.T), f16 in / f32 accumulate.
// Block = 128 threads (4 waves, 2x2), block tile 64x128, wave tile 32x64
// (2x4 WMMA accumulators, 8 WMMAs per K=32 step from 6 fragment loads).
// MODE 0: f16 row-major out, scaled by outScale (Q gets F^-0.5 folded in)
// MODE 1: f16 transposed out  Vt[b,h,d,s]  (V projection)
// MODE 2: f32 out, (C + bias) * (1 - dw)   (output projection)
// ---------------------------------------------------------------------------
template <int MODE>
__global__ __launch_bounds__(128, 1) void gemm_xWT_f16_wmma(
    const _Float16* __restrict__ A, const _Float16* __restrict__ W,
    int M, int N, int K,
    _Float16* __restrict__ outh, float* __restrict__ outf,
    const float* __restrict__ bias, const float* __restrict__ dwp,
    float outScale) {
  const int lane = threadIdx.x & 31;
  const int wave = threadIdx.x >> 5;
  const int hi = lane >> 4, ln = lane & 15;
  const int m0 = blockIdx.y * 64 + (wave >> 1) * 32;
  const int n0 = blockIdx.x * 128 + (wave & 1) * 64;

  v8f acc[2][4] = {};

  for (int k0 = 0; k0 < K; k0 += 32) {
    if (k0 + 32 < K) {
      __builtin_prefetch((const void*)(A + (m0 + ln) * K + k0 + 32), 0, 3);
      __builtin_prefetch((const void*)(W + (n0 + ln) * K + k0 + 32), 0, 3);
    }
    v16h af[2], bf[4];
#pragma unroll
    for (int i = 0; i < 2; i++) {
      const _Float16* ra = A + (m0 + i * 16 + ln) * K + k0;
      af[i] = frag_from2(ra + 16 * hi, ra + 16 * hi + 8);
    }
#pragma unroll
    for (int j = 0; j < 4; j++) {
      const _Float16* rb = W + (n0 + j * 16 + ln) * K + k0 + 16 * hi;
      bf[j] = frag_from2(rb, rb + 8);
    }
#pragma unroll
    for (int i = 0; i < 2; i++)
#pragma unroll
      for (int j = 0; j < 4; j++)
        acc[i][j] = wmma_f16(af[i], bf[j], acc[i][j]);
  }

  float sc = 1.0f;
  if (MODE == 2) sc = 1.0f - *dwp;
#pragma unroll
  for (int i = 0; i < 2; i++) {
#pragma unroll
    for (int j = 0; j < 4; j++) {
      const int mbase = m0 + i * 16 + hi * 8;  // 8 consecutive rows (tokens)
      const int n = n0 + j * 16 + ln;
      if (MODE == 0) {
#pragma unroll
        for (int v = 0; v < 8; v++)
          outh[(mbase + v) * N + n] = (_Float16)(acc[i][j][v] * outScale);
      } else if (MODE == 1) {
        // Vt[b, h, d, s]: 8 consecutive s at fixed d -> one contiguous v8h
        const int hh = n >> 7, d = n & 127;
        const int bb = mbase >> 11, s = mbase & 2047;
        v8h pack;
#pragma unroll
        for (int v = 0; v < 8; v++) pack[v] = (_Float16)acc[i][j][v];
        *(v8h*)(outh + ((bb * HHH + hh) * DHD + d) * SSQ + s) = pack;
      } else {
        const float bb = bias[n];
#pragma unroll
        for (int v = 0; v < 8; v++)
          outf[(mbase + v) * N + n] = (acc[i][j][v] + bb) * sc;
      }
    }
  }
}

// ---------------------------------------------------------------------------
// Differential flash attention. Block = 64 threads (2 waves); wave `br` owns
// softmax branch `br`. One block per (b, h, 16-row q tile).
// Scores are pre-scaled (F^-0.5 folded into the Q projection).
// Softmax: scatter score tile to LDS, each lane owns one half-row (row=ln,
// half=hi), reduces 16 values serially in registers, one shfl_xor(16) merges
// the halves. Per-row m/l state is a single scalar per lane.
// ---------------------------------------------------------------------------
#define SLPAD 36  // f32 row stride for score tile (bank-conflict padding)
#define PLPAD 40  // f16 row stride for P tile

__global__ __launch_bounds__(64, 1) void diffattn_kernel(
    const _Float16* __restrict__ Qh, const _Float16* __restrict__ Kh,
    const _Float16* __restrict__ Vt, float* __restrict__ O,
    const float* __restrict__ dwp) {
  const int qt = blockIdx.x, h = blockIdx.y, b = blockIdx.z;
  const int br = threadIdx.x >> 5;  // wave id == branch id
  const int lane = threadIdx.x & 31;
  const int hi = lane >> 4, ln = lane & 15;
  const int q0 = qt * 16;
  const int bs = b * SSQ;

  __shared__ float    Sl[2][16][SLPAD];   // score tiles
  __shared__ _Float16 Pl[2][16][PLPAD];   // P tiles
  __shared__ float    corrBuf[2][16];
  __shared__ float    linvBuf[2][16];
  __shared__ float    Comb[16][DHD];      // branch-1 contribution

  // Resident Q A-fragments for this branch: [kf]
  v16h qA[2];
#pragma unroll
  for (int kf = 0; kf < 2; kf++) {
    const _Float16* r =
        Qh + (bs + q0 + ln) * EEE + h * DHD + br * FFF + kf * 32;
    qA[kf] = frag_from2(r + 16 * hi, r + 16 * hi + 8);
  }

  const _Float16* VtB = Vt + (b * HHH + h) * DHD * SSQ;  // + d*SSQ + s

  v8f accO[8] = {};
  float m_r = -INFINITY;  // running max for row `ln` (both halves agree)
  float l_r = 0.f;        // running denom for row `ln`

  for (int k0 = 0; k0 < SSQ; k0 += 32) {
    if (k0 + 32 < SSQ)
      __builtin_prefetch(
          (const void*)(Kh + (bs + k0 + 32 + ln) * EEE + h * DHD + br * FFF),
          0, 3);

    // ---- scores S = Q @ K^T : [nsub] 16x16 tiles (pre-scaled) ----
    v8f s[2] = {};
#pragma unroll
    for (int ns = 0; ns < 2; ns++)
#pragma unroll
      for (int kf = 0; kf < 2; kf++) {
        const _Float16* r = Kh + (bs + k0 + ns * 16 + ln) * EEE + h * DHD +
                            br * FFF + kf * 32 + 16 * hi;
        s[ns] = wmma_f16(qA[kf], frag_from2(r, r + 8), s[ns]);
      }

    // ---- scatter scores to LDS (C-layout -> row-major tile) ----
#pragma unroll
    for (int v = 0; v < 8; v++) {
      Sl[br][hi * 8 + v][ln] = s[0][v];
      Sl[br][hi * 8 + v][16 + ln] = s[1][v];
    }

    // ---- half-row gather + online softmax (row = ln, half = hi) ----
    const float4* rp = (const float4*)&Sl[br][ln][hi * 16];
    float4 c0 = rp[0], c1 = rp[1], c2 = rp[2], c3 = rp[3];
    float e[16] = {c0.x, c0.y, c0.z, c0.w, c1.x, c1.y, c1.z, c1.w,
                   c2.x, c2.y, c2.z, c2.w, c3.x, c3.y, c3.z, c3.w};
    float mx = e[0];
#pragma unroll
    for (int j = 1; j < 16; j++) mx = fmaxf(mx, e[j]);
    mx = fmaxf(mx, __shfl_xor(mx, 16, 32));  // merge the two half-rows
    const float mnew = fmaxf(m_r, mx);
    const float corr = __expf(m_r - mnew);
    float sum = 0.f;
    v8h p0, p1;
#pragma unroll
    for (int j = 0; j < 8; j++) {
      const float pa = __expf(e[j] - mnew);
      const float pb = __expf(e[8 + j] - mnew);
      sum += pa + pb;
      p0[j] = (_Float16)pa;
      p1[j] = (_Float16)pb;
    }
    sum += __shfl_xor(sum, 16, 32);
    l_r = l_r * corr + sum;
    m_r = mnew;

    // P row back to LDS (contiguous 32B per half-row)
    *(v8h*)&Pl[br][ln][hi * 16] = p0;
    *(v8h*)&Pl[br][ln][hi * 16 + 8] = p1;
    if (hi == 0) corrBuf[br][ln] = corr;

    // ---- rescale O accumulators by this tile's corr factors ----
    const float4* cb = (const float4*)&corrBuf[br][hi * 8];
    float4 ca = cb[0], cc = cb[1];
    const float cw[8] = {ca.x, ca.y, ca.z, ca.w, cc.x, cc.y, cc.z, cc.w};
#pragma unroll
    for (int d = 0; d < 8; d++)
#pragma unroll
      for (int v = 0; v < 8; v++) accO[d][v] *= cw[v];

    // ---- O += P @ V (B-fragments contiguous from transposed Vt) ----
    const _Float16* pr = &Pl[br][ln][0];
    v16h pA = frag_from2(pr + 16 * hi, pr + 16 * hi + 8);
#pragma unroll
    for (int d = 0; d < 8; d++) {
      const _Float16* vr = VtB + (d * 16 + ln) * SSQ + k0 + 16 * hi;
      accO[d] = wmma_f16(pA, frag_from2(vr, vr + 8), accO[d]);
    }
  }

  // ---- combine branches: O = O0/l0 - dw * O1/l1 ----
  if (hi == 0) linvBuf[br][ln] = 1.0f / l_r;
  const float4* lb = (const float4*)&linvBuf[br][hi * 8];
  float4 la = lb[0], lc = lb[1];
  const float lw[8] = {la.x, la.y, la.z, la.w, lc.x, lc.y, lc.z, lc.w};

  const float dw = *dwp;
  if (br == 1) {
#pragma unroll
    for (int d = 0; d < 8; d++)
#pragma unroll
      for (int v = 0; v < 8; v++)
        Comb[hi * 8 + v][d * 16 + ln] = dw * accO[d][v] * lw[v];
  }
  __syncthreads();
  if (br == 0) {
#pragma unroll
    for (int d = 0; d < 8; d++)
#pragma unroll
      for (int v = 0; v < 8; v++) {
        const int q = q0 + hi * 8 + v;
        O[(bs + q) * EEE + h * DHD + d * 16 + ln] =
            accO[d][v] * lw[v] - Comb[hi * 8 + v][d * 16 + ln];
      }
  }
}

// ---------------------------------------------------------------------------
// RMSNorm over the full 1024-wide token row, write f16 for the final GEMM.
// ---------------------------------------------------------------------------
__global__ void rmsnorm_kernel(const float* __restrict__ O,
                               const float* __restrict__ norm_w,
                               _Float16* __restrict__ Nh) {
  const int t = blockIdx.x;
  const float* row = O + (size_t)t * EEE;
  __shared__ float red[256];
  float ss = 0.f;
  for (int i = threadIdx.x; i < EEE; i += 256) {
    const float v = row[i];
    ss += v * v;
  }
  red[threadIdx.x] = ss;
  __syncthreads();
  for (int s = 128; s > 0; s >>= 1) {
    if ((int)threadIdx.x < s) red[threadIdx.x] += red[threadIdx.x + s];
    __syncthreads();
  }
  const float rms = rsqrtf(red[0] * (1.0f / EEE) + 1.1920929e-07f);
  for (int i = threadIdx.x; i < EEE; i += 256)
    Nh[(size_t)t * EEE + i] = (_Float16)(row[i] * rms * norm_w[i]);
}

// ---------------------------------------------------------------------------
extern "C" void kernel_launch(void* const* d_in, const int* in_sizes, int n_in,
                              void* d_out, int out_size, void* d_ws,
                              size_t ws_size, hipStream_t stream) {
  (void)in_sizes; (void)n_in; (void)out_size; (void)ws_size;
  const float* x  = (const float*)d_in[0];
  const float* Wq = (const float*)d_in[1];
  const float* Wk = (const float*)d_in[2];
  const float* Wv = (const float*)d_in[3];
  const float* nw = (const float*)d_in[4];
  const float* Wo = (const float*)d_in[5];
  const float* bo = (const float*)d_in[6];
  const float* dw = (const float*)d_in[7];

  char* ws = (char*)d_ws;
  _Float16* xh  = (_Float16*)(ws);                       //  8 MB
  _Float16* Wqh = (_Float16*)(ws + ((size_t)8  << 20));  //  2 MB
  _Float16* Wkh = (_Float16*)(ws + ((size_t)10 << 20));  //  2 MB
  _Float16* Wvh = (_Float16*)(ws + ((size_t)12 << 20));  //  2 MB
  _Float16* Woh = (_Float16*)(ws + ((size_t)14 << 20));  //  2 MB
  _Float16* Qh  = (_Float16*)(ws + ((size_t)16 << 20));  //  8 MB
  _Float16* Kh  = (_Float16*)(ws + ((size_t)24 << 20));  //  8 MB
  _Float16* Vt  = (_Float16*)(ws + ((size_t)32 << 20));  //  8 MB [b,h,d,s]
  float*    Of  = (float*)   (ws + ((size_t)40 << 20));  // 16 MB
  _Float16* Nh  = (_Float16*)(ws + ((size_t)56 << 20));  //  8 MB

  const int M = BB * SSQ;       // 4096
  const int xn8 = M * EEE / 8;
  const int wn8 = EEE * EEE / 8;

  cast_f32_to_f16_kernel<<<(xn8 + 255) / 256, 256, 0, stream>>>(x, xh, xn8);
  cast_f32_to_f16_kernel<<<(wn8 + 255) / 256, 256, 0, stream>>>(Wq, Wqh, wn8);
  cast_f32_to_f16_kernel<<<(wn8 + 255) / 256, 256, 0, stream>>>(Wk, Wkh, wn8);
  cast_f32_to_f16_kernel<<<(wn8 + 255) / 256, 256, 0, stream>>>(Wv, Wvh, wn8);
  cast_f32_to_f16_kernel<<<(wn8 + 255) / 256, 256, 0, stream>>>(Wo, Woh, wn8);

  dim3 ggrid(EEE / 128, M / 64);
  const float qscale = 0.125f;  // F^-0.5 folded into Q projection
  gemm_xWT_f16_wmma<0><<<ggrid, 128, 0, stream>>>(
      xh, Wqh, M, EEE, EEE, Qh, nullptr, nullptr, nullptr, qscale);
  gemm_xWT_f16_wmma<0><<<ggrid, 128, 0, stream>>>(
      xh, Wkh, M, EEE, EEE, Kh, nullptr, nullptr, nullptr, 1.0f);
  gemm_xWT_f16_wmma<1><<<ggrid, 128, 0, stream>>>(
      xh, Wvh, M, EEE, EEE, Vt, nullptr, nullptr, nullptr, 1.0f);

  diffattn_kernel<<<dim3(SSQ / 16, HHH, BB), 64, 0, stream>>>(Qh, Kh, Vt, Of,
                                                              dw);

  rmsnorm_kernel<<<M, 256, 0, stream>>>(Of, nw, Nh);

  gemm_xWT_f16_wmma<2><<<ggrid, 128, 0, stream>>>(
      Nh, Woh, M, EEE, EEE, nullptr, (float*)d_out, bo, dw, 1.0f);
}
